// Bahdanau_Attention_32865089749661
// MI455X (gfx1250) — matrix operand order repaired and checked by reading, not compiled, not run on hardware
//
#include <hip/hip_runtime.h>
#include <math.h>

// ---------------------------------------------------------------------------
// Bahdanau attention, fused for MI455X (gfx1250, wave32, WMMA).
//   B=32, S=2048, H=1024 (fp32 in/out; GEMM in bf16 w/ f32 accumulate)
// Round 2: 32-row M tile per block -> each B fragment from L2 feeds 2 WMMAs
// (halves L2 traffic of the weight operand; HBM traffic unchanged).
// ---------------------------------------------------------------------------

#define Hdim 1024
#define Sdim 2048
#define Bdim 32

typedef __attribute__((ext_vector_type(16))) __bf16       v16bf;
typedef __attribute__((ext_vector_type(4)))  __bf16       v4bf;
typedef __attribute__((ext_vector_type(8)))  float        v8f;
typedef __attribute__((ext_vector_type(4)))  float        f32x4;
typedef __attribute__((ext_vector_type(4)))  unsigned int u32x4;

union Frag { u32x4 q[2]; v16bf v; };

// ---------------------------------------------------------------------------
// Kernel 1: WeT[n][k] = (bf16) We[k][n]   (transposed so B-fragments are
// contiguous 16-byte runs per lane). 2 MB result -> lives in L2 for the GEMM.
// grid: (H), block: 256
// ---------------------------------------------------------------------------
__global__ __launch_bounds__(256)
void prep_weights_kernel(const float* __restrict__ We, __bf16* __restrict__ wet) {
  const int n = blockIdx.x;
  const int tid = threadIdx.x;
#pragma unroll
  for (int i = 0; i < 4; ++i) {
    const int k = i * 256 + tid;
    wet[(size_t)n * Hdim + k] = (__bf16)We[(size_t)k * Hdim + n];
  }
}

// ---------------------------------------------------------------------------
// Kernel 2: biasc[b][n] = be[n] + bd[n] + sum_k dec[b][k] * Wd[k][n]
// grid: (H/256, B), block: 256   (Wd reads coalesced in n)
// ---------------------------------------------------------------------------
__global__ __launch_bounds__(256)
void prep_bias_kernel(const float* __restrict__ dhs, const float* __restrict__ Wd,
                      const float* __restrict__ bd, const float* __restrict__ be,
                      float* __restrict__ biasc) {
  const int b = blockIdx.y;
  const int n = blockIdx.x * 256 + threadIdx.x;
  float acc = bd[n] + be[n];
  const float* __restrict__ d = dhs + (size_t)b * Hdim;
#pragma unroll 4
  for (int k = 0; k < Hdim; ++k)
    acc = fmaf(d[k], Wd[(size_t)k * Hdim + n], acc);
  biasc[(size_t)b * Hdim + n] = acc;
}

// ---------------------------------------------------------------------------
// Kernel 3: fused  linear_score[b][s] =
//     tanh(enc[b][s][:] @ We + biasc[b][:]) @ Ws + bs
// One block per (b, 32-row S tile). 8 waves; wave w owns N-tiles 8w..8w+7 for
// BOTH 16-row M tiles (B fragment reused x2). A tiles (32 x 1024) staged in
// LDS as bf16 (64 KB); 32 K-steps of v_wmma_f32_16x16x32_bf16 per N-tile.
// grid: (S/32, B), block: 256
// ---------------------------------------------------------------------------
__global__ __launch_bounds__(256)
void fused_score_kernel(const float*  __restrict__ enc,
                        const __bf16* __restrict__ wet,
                        const float*  __restrict__ biasc,
                        const float*  __restrict__ Ws,
                        const float*  __restrict__ bs,
                        float*        __restrict__ lscore) {
  __shared__ __align__(16) __bf16 a_lds[32 * Hdim];   // 64 KB
  __shared__ float row_sum[32];

  const int tid = threadIdx.x;
  const int b   = blockIdx.y;
  const int s0  = blockIdx.x * 32;

  // --- cooperative load + fp32->bf16 convert of the A tile (contiguous) ---
  const f32x4* __restrict__ encv =
      (const f32x4*)(enc + ((size_t)b * Sdim + s0) * Hdim);
#pragma unroll
  for (int i = 0; i < 32; ++i) {
    const int j = i * 256 + tid;              // 8192 f32x4 total
    f32x4 x = encv[j];
    v4bf y = { (__bf16)x[0], (__bf16)x[1], (__bf16)x[2], (__bf16)x[3] };
    *(v4bf*)(a_lds + (size_t)j * 4) = y;
  }
  if (tid < 32) row_sum[tid] = 0.0f;
  __syncthreads();

  const int wave = tid >> 5;
  const int lane = tid & 31;
  const int half = lane >> 4;   // which 16-lane half
  const int ln   = lane & 15;

  const v8f zero = {0.f, 0.f, 0.f, 0.f, 0.f, 0.f, 0.f, 0.f};
  v8f acc0[8], acc1[8];
#pragma unroll
  for (int t = 0; t < 8; ++t) { acc0[t] = zero; acc1[t] = zero; }

  // A-fragment rows: lane ln -> matrix rows ln (tile 0) and 16+ln (tile 1).
  const __bf16* __restrict__ arow0 = a_lds + (size_t)ln * Hdim;
  const __bf16* __restrict__ arow1 = a_lds + (size_t)(16 + ln) * Hdim;

  for (int kk = 0; kk < Hdim; kk += 32) {
    // 16-bit A 16x32 layout: lane holds K = [kk+8h, kk+8h+8) ++ [kk+16+8h, kk+24+8h)
    Frag a0, a1;
    a0.q[0] = *(const u32x4*)(arow0 + kk + 8 * half);
    a0.q[1] = *(const u32x4*)(arow0 + kk + 16 + 8 * half);
    a1.q[0] = *(const u32x4*)(arow1 + kk + 8 * half);
    a1.q[1] = *(const u32x4*)(arow1 + kk + 16 + 8 * half);
#pragma unroll
    for (int t = 0; t < 8; ++t) {
      const int n = ((wave * 8 + t) << 4) + ln;        // output column
      // 16-bit B 32x16 layout: lane = column n, K = [kk+8h, kk+8h+8) ++ [kk+16+8h, kk+24+8h)
      const __bf16* __restrict__ bp = wet + (size_t)n * Hdim + kk + 8 * half;
      Frag bm;
      bm.q[0] = ((const u32x4*)bp)[0];
      bm.q[1] = ((const u32x4*)bp)[2];
      acc0[t] = __builtin_amdgcn_wmma_f32_16x16x32_bf16(
          false, a0.v, false, bm.v, (short)0, acc0[t], false, false);
      acc1[t] = __builtin_amdgcn_wmma_f32_16x16x32_bf16(
          false, a1.v, false, bm.v, (short)0, acc1[t], false, false);
    }
  }

  // --- epilogue: +bias, tanh, *Ws[n], reduce over n into per-row scalars ---
  // C layout: VGPR r, lane -> M = r + 8*half (+16 for tile 1), N = ln.
  float p0[8] = {0.f, 0.f, 0.f, 0.f, 0.f, 0.f, 0.f, 0.f};
  float p1[8] = {0.f, 0.f, 0.f, 0.f, 0.f, 0.f, 0.f, 0.f};
#pragma unroll
  for (int t = 0; t < 8; ++t) {
    const int n = ((wave * 8 + t) << 4) + ln;
    const float bias = biasc[(size_t)b * Hdim + n];
    const float wsn  = Ws[n];
#pragma unroll
    for (int r = 0; r < 8; ++r) {
      p0[r] = fmaf(tanhf(acc0[t][r] + bias), wsn, p0[r]);
      p1[r] = fmaf(tanhf(acc1[t][r] + bias), wsn, p1[r]);
    }
  }
#pragma unroll
  for (int r = 0; r < 8; ++r) {
    atomicAdd(&row_sum[r + 8 * half], p0[r]);        // ds_add_f32
    atomicAdd(&row_sum[16 + r + 8 * half], p1[r]);
  }

  __syncthreads();
  if (tid < 32)
    lscore[(size_t)b * Sdim + s0 + tid] = row_sum[tid] + bs[0];
}

// ---------------------------------------------------------------------------
// Kernel 4: softmax over S per batch, in place. grid: (B), block: 256
// ---------------------------------------------------------------------------
__global__ __launch_bounds__(256)
void softmax_kernel(float* __restrict__ sc) {
  __shared__ float red[256];
  const int b = blockIdx.x, tid = threadIdx.x;
  float* __restrict__ row = sc + (size_t)b * Sdim;

  float v[8];
  float m = -3.4e38f;
#pragma unroll
  for (int i = 0; i < 8; ++i) { v[i] = row[tid + i * 256]; m = fmaxf(m, v[i]); }
  red[tid] = m; __syncthreads();
  for (int st = 128; st > 0; st >>= 1) {
    if (tid < st) red[tid] = fmaxf(red[tid], red[tid + st]);
    __syncthreads();
  }
  m = red[0]; __syncthreads();

  float s = 0.f;
#pragma unroll
  for (int i = 0; i < 8; ++i) { v[i] = __expf(v[i] - m); s += v[i]; }
  red[tid] = s; __syncthreads();
  for (int st = 128; st > 0; st >>= 1) {
    if (tid < st) red[tid] += red[tid + st];
    __syncthreads();
  }
  const float inv = 1.0f / red[0];
#pragma unroll
  for (int i = 0; i < 8; ++i) row[tid + i * 256] = v[i] * inv;
}

// ---------------------------------------------------------------------------
// Kernel 5: context[b][h] = sum_s attn[b][s] * enc[b][s][h]
// Pure-bandwidth second pass over encoder_out (coalesced in h).
// grid: (H/256, B), block: 256
// ---------------------------------------------------------------------------
__global__ __launch_bounds__(256)
void context_kernel(const float* __restrict__ enc, const float* __restrict__ attn,
                    float* __restrict__ out) {
  const int b = blockIdx.y;
  const int h = blockIdx.x * 256 + threadIdx.x;
  const float* __restrict__ eb = enc + (size_t)b * Sdim * Hdim + h;
  const float* __restrict__ ab = attn + (size_t)b * Sdim;
  float a0 = 0.f, a1 = 0.f, a2 = 0.f, a3 = 0.f;
  for (int s = 0; s < Sdim; s += 4) {
    a0 = fmaf(ab[s + 0], eb[(size_t)(s + 0) * Hdim], a0);
    a1 = fmaf(ab[s + 1], eb[(size_t)(s + 1) * Hdim], a1);
    a2 = fmaf(ab[s + 2], eb[(size_t)(s + 2) * Hdim], a2);
    a3 = fmaf(ab[s + 3], eb[(size_t)(s + 3) * Hdim], a3);
  }
  out[(size_t)b * Hdim + h] = (a0 + a1) + (a2 + a3);
}

// ---------------------------------------------------------------------------
// Launch. Inputs (setup_inputs order):
//  0 encoder_out [32,2048,1024] f32   1 decoder_hidden_state [1,32,1024] f32
//  2 We [1024,1024] f32  3 be [1024]  4 Wd [1024,1024]  5 bd [1024]
//  6 Ws [1024,1]         7 bs [1]
// Output: context_vector [32,1024] f32
// Workspace: WeT bf16 (2MB) | biasc (128KB) | linear_score/attn (256KB)
// ---------------------------------------------------------------------------
extern "C" void kernel_launch(void* const* d_in, const int* in_sizes, int n_in,
                              void* d_out, int out_size, void* d_ws, size_t ws_size,
                              hipStream_t stream) {
  const float* enc = (const float*)d_in[0];
  const float* dhs = (const float*)d_in[1];
  const float* We  = (const float*)d_in[2];
  const float* be  = (const float*)d_in[3];
  const float* Wd  = (const float*)d_in[4];
  const float* bd  = (const float*)d_in[5];
  const float* Ws  = (const float*)d_in[6];
  const float* bs  = (const float*)d_in[7];
  float* out = (float*)d_out;

  char* ws = (char*)d_ws;
  __bf16* wet   = (__bf16*)ws;                                   // 2 MB
  float*  biasc = (float*)(ws + (size_t)2 * 1024 * 1024);        // 128 KB
  float*  lsc   = (float*)(ws + (size_t)2 * 1024 * 1024 + 128 * 1024); // 256 KB

  prep_weights_kernel<<<dim3(Hdim), dim3(256), 0, stream>>>(We, wet);
  prep_bias_kernel<<<dim3(Hdim / 256, Bdim), dim3(256), 0, stream>>>(dhs, Wd, bd, be, biasc);
  fused_score_kernel<<<dim3(Sdim / 32, Bdim), dim3(256), 0, stream>>>(enc, wet, biasc, Ws, bs, lsc);
  softmax_kernel<<<dim3(Bdim), dim3(256), 0, stream>>>(lsc);
  context_kernel<<<dim3(Hdim / 256, Bdim), dim3(256), 0, stream>>>(enc, lsc, out);
}
